// Model_44770739093726
// MI455X (gfx1250) — matrix operand order repaired and checked by reading, not compiled, not run on hardware
//
#include <hip/hip_runtime.h>

typedef __attribute__((ext_vector_type(16))) _Float16 v16h;
typedef __attribute__((ext_vector_type(8)))  _Float16 v8h;
typedef __attribute__((ext_vector_type(8)))  float    v8f;

// Problem constants (from reference)
constexpr float kBeta = 0.9f;
constexpr float kThr  = 0.5f;
constexpr int   kT    = 50;
constexpr int   kB    = 131072;

// Tiling
constexpr int kWavesPerBlock = 8;
constexpr int kBatchPerWave  = 16;
constexpr int kBlockThreads  = kWavesPerBlock * 32;
constexpr int kBatchPerBlock = kWavesPerBlock * kBatchPerWave;

__global__ __launch_bounds__(kBlockThreads, 1)
void snn_lif_wmma(const float* __restrict__ x,
                  const float* __restrict__ W1,
                  const float* __restrict__ W2,
                  float* __restrict__ out)
{
    // Padded weights staged in LDS.
    // W1p: [128][2] f32 (hidden padded 100->128), zeros beyond.
    // W2p: [16][128] f16 (outputs padded 2->16, hidden padded 100->128), zeros beyond.
    __shared__ __align__(16) float    W1p[128 * 2];
    __shared__ __align__(16) _Float16 W2p[16 * 128];

    const int tid = threadIdx.x;
    for (int k = tid; k < 256; k += kBlockThreads)
        W1p[k] = (k < 200) ? W1[k] : 0.0f;
    for (int k = tid; k < 2048; k += kBlockThreads) {
        const int rr = k >> 7, cc = k & 127;
        W2p[k] = (rr < 2 && cc < 100) ? (_Float16)W2[rr * 100 + cc] : (_Float16)0.0f;
    }
    __syncthreads();

    const int lane = tid & 31;
    const int wave = tid >> 5;
    const int r = lane & 15;   // batch row within tile / A-matrix row
    const int h = lane >> 4;   // lane half (selects K sub-group in WMMA layouts)
    const int batchBase = (blockIdx.x * kWavesPerBlock + wave) * kBatchPerWave;
    const int b = batchBase + r;

    // Preload A = W2 (padded, f16) in WMMA A-matrix layout, 4 chunks of K=32.
    // K order used here (contiguous, same as B): lanes 0-15 row M=lane hold K = 16h..16h+7 (V0-3) and
    // 16h+8..16h+15 (V4-7).
    v16h A[4];
    #pragma unroll
    for (int c = 0; c < 4; ++c) {
        const v8h lo = *(const v8h*)&W2p[r * 128 + c * 32 + 16 * h];
        const v8h hi = *(const v8h*)&W2p[r * 128 + c * 32 + 16 * h + 8];
        #pragma unroll
        for (int q = 0; q < 8; ++q) { A[c][q] = lo[q]; A[c][8 + q] = hi[q]; }
    }

    // mem1 state in registers, arranged as WMMA B-matrix (K x N) layout:
    // lane owns batch n = lane%16; chunk c, half h -> j = 32c + 16h + i, i=0..15.
    float mem1[64];
    #pragma unroll
    for (int i = 0; i < 64; ++i) mem1[i] = 0.0f;
    float mem20 = 0.0f, mem21 = 0.0f;

    const size_t memOff = (size_t)kT * kB * 2;   // offset of mem_rec in d_out

    for (int t = 0; t < kT; ++t) {
        const float2 xt = *(const float2*)(x + ((size_t)t * kB + b) * 2);
        if (t + 1 < kT)
            __builtin_prefetch(x + ((size_t)(t + 1) * kB + b) * 2, 0, 3);
        const float x0 = xt.x, x1 = xt.y;

        v8f acc = {};   // cur2 accumulator (C starts at zero each step)
        #pragma unroll
        for (int c = 0; c < 4; ++c) {
            const int J = c * 32 + 16 * h;   // this lane's base hidden index
            v16h s;                           // spikes, B-matrix layout chunk
            #pragma unroll
            for (int v = 0; v < 8; ++v) {
                // W1 rows for j pair (J+2v, J+2v+1): 16 aligned bytes in LDS
                const float4 w = *(const float4*)&W1p[(J + 2 * v) * 2];
                const int i0 = c * 16 + 2 * v;
                float m0 = mem1[i0], m1 = mem1[i0 + 1];
                // reset-by-subtraction uses the PREVIOUS membrane (== prev spike)
                const float r0 = (m0 > kThr) ? -kThr : 0.0f;
                const float r1 = (m1 > kThr) ? -kThr : 0.0f;
                m0 = __builtin_fmaf(kBeta, m0, r0);
                m0 = __builtin_fmaf(x0, w.x, m0);
                m0 = __builtin_fmaf(x1, w.y, m0);
                m1 = __builtin_fmaf(kBeta, m1, r1);
                m1 = __builtin_fmaf(x0, w.z, m1);
                m1 = __builtin_fmaf(x1, w.w, m1);
                mem1[i0] = m0; mem1[i0 + 1] = m1;
                s[2 * v]     = (m0 > kThr) ? (_Float16)1.0f : (_Float16)0.0f;
                s[2 * v + 1] = (m1 > kThr) ? (_Float16)1.0f : (_Float16)0.0f;
            }
            // cur2 += W2_chunk (16x32) * spikes_chunk (32x16)
            acc = __builtin_amdgcn_wmma_f32_16x16x32_f16(
                false, A[c], false, s, (short)0, acc, false, false);
        }

        // D layout: VGPR0 lanes 0-15 = row M=0 (out0), VGPR1 = row M=1 (out1),
        // N = lane. Lanes 16-31 hold zero-padded rows (harmless).
        const float cur20 = acc[0], cur21 = acc[1];
        const float rs0 = (mem20 > kThr) ? kThr : 0.0f;
        const float rs1 = (mem21 > kThr) ? kThr : 0.0f;
        mem20 = __builtin_fmaf(kBeta, mem20, cur20) - rs0;
        mem21 = __builtin_fmaf(kBeta, mem21, cur21) - rs1;
        const float spk20 = (mem20 > kThr) ? 1.0f : 0.0f;
        const float spk21 = (mem21 > kThr) ? 1.0f : 0.0f;

        if (h == 0) {   // lanes 0-15 own the real outputs; reconverges each iter
            float2 s2; s2.x = spk20; s2.y = spk21;
            float2 m2; m2.x = mem20; m2.y = mem21;
            *(float2*)(out + ((size_t)t * kB + b) * 2)          = s2;
            *(float2*)(out + memOff + ((size_t)t * kB + b) * 2) = m2;
        }
    }
}

extern "C" void kernel_launch(void* const* d_in, const int* in_sizes, int n_in,
                              void* d_out, int out_size, void* d_ws, size_t ws_size,
                              hipStream_t stream) {
    const float* x  = (const float*)d_in[0];   // [50,131072,2] f32
    const float* W1 = (const float*)d_in[1];   // [100,2] f32
    const float* W2 = (const float*)d_in[2];   // [2,100] f32
    float* out = (float*)d_out;                // spk_rec ++ mem_rec, each [50,131072,2]

    dim3 grid(kB / kBatchPerBlock);            // 1024 blocks
    dim3 block(kBlockThreads);                 // 256 threads = 8 waves
    hipLaunchKernelGGL(snn_lif_wmma, grid, block, 0, stream, x, W1, W2, out);
}
